// NewGNNEncoder_58463094833683
// MI455X (gfx1250) — compile-verified, matrix operand order checked by reading, not run on hardware
//
#include <hip/hip_runtime.h>
#include <stdint.h>

typedef __bf16 bf16_t;
typedef __attribute__((ext_vector_type(16))) __bf16 v16bf;
typedef __attribute__((ext_vector_type(8)))  __bf16 bf16x8;
typedef __attribute__((ext_vector_type(8)))  float  v8f;

#define HIDN   64
#define NCHN   4
#define ATTRD  35
#define RADK   1225
#define RADKP  1248   // padded to multiple of 32 for WMMA K-chunks
#define EPN    70     // edges emitted per node: 30+10+15+15
#define MAXB   2048

// dynamic-LDS weight stage layout (elements)
#define WE1_ELEMS (192*64)
#define WE2_ELEMS (64*64)
#define WC1_ELEMS (64*64)
#define WSTAGE_ELEMS (WE1_ELEMS + WE2_ELEMS + WC1_ELEMS)   // 20480 bf16 = 40960 B

// ---------------- numeric helpers ----------------
__device__ __forceinline__ float sigmoidf_(float x){ return 1.f/(1.f+__expf(-x)); }
__device__ __forceinline__ float siluf_(float x){ return x*sigmoidf_(x); }
__device__ __forceinline__ bf16_t f2bf(float f){
  union { float f; uint32_t u; } a; a.f=f;
  uint16_t h=(uint16_t)((a.u + 0x7FFFu + ((a.u>>16)&1u))>>16);
  union { uint16_t u; bf16_t b; } c; c.u=h;
  return c.b;
}

// ---------------- gfx1250 async copy helpers ----------------
// LDS byte address = low 32 bits of the generic (flat) address (ISA 10.2: LDS
// aperture truncates addr[31:0]).  16B per lane per issue; tracked by ASYNCcnt.
__device__ __forceinline__ void async_copy_b128(void* ldsDst, const void* gsrc){
  uint32_t l = (uint32_t)(uintptr_t)ldsDst;
  asm volatile("global_load_async_to_lds_b128 %0, %1, off"
               :: "v"(l), "v"(gsrc) : "memory");
}
__device__ __forceinline__ void wait_async0(){
  asm volatile("s_wait_asynccnt 0x0" ::: "memory");
}

// ---------------- WMMA tile helpers (wave32, 16x16x32 bf16) ----------------
// A staged in LDS as [16 rows][KP] bf16, KP multiple of 32, rows 16B aligned.
// A-fragment: lane l: m=l&15, g=l>>4; elems 0..7 = K k0+g*8.. ; 8..15 = K k0+16+g*8..
__device__ __forceinline__ v16bf ld_a(const bf16_t* Abuf, int KP, int k0, int lane){
  int m = lane & 15, g = lane >> 4;
  const bf16_t* p = Abuf + m*KP + k0 + g*8;
  union { v16bf v; bf16x8 h[2]; } u;
  u.h[0] = *(const bf16x8*)p;
  u.h[1] = *(const bf16x8*)(p + 16);
  return u.v;
}
// B pre-packed so each lane's 16 elems are contiguous (32B): one b128 pair.
__device__ __forceinline__ v16bf ld_b(const bf16_t* p){
  union { v16bf v; bf16x8 h[2]; } u;
  u.h[0] = *(const bf16x8*)p;
  u.h[1] = *(const bf16x8*)(p + 8);
  return u.v;
}
// one wave computes a 16(M) x 16(N) tile over full K; NT fixed = 4 (Nout=64)
__device__ __forceinline__ v8f gemm16(const bf16_t* Abuf, const bf16_t* Bpack,
                                      int KP, int nt, int lane){
  v8f acc = {};
  int chunk = 0;
  for (int k0 = 0; k0 < KP; k0 += 32, ++chunk){
    v16bf a = ld_a(Abuf, KP, k0, lane);
    const bf16_t* bp = Bpack + (((size_t)chunk*4 + nt)*32 + lane)*16;
    if (KP > 256 && k0 + 64 < KP)   // long weight stream (radial): prefetch 2 chunks ahead
      __builtin_prefetch(bp + (size_t)2*4*32*16, 0, 1);
    v16bf b = ld_b(bp);
    acc = __builtin_amdgcn_wmma_f32_16x16x32_bf16(false, a, false, b, (short)0, acc, false, false);
  }
  return acc;
}

// ---------------- weight pre-pack: W (Kdim x 64, row-major f32) -> B fragments ----------------
__global__ void k_pack(const float* __restrict__ W, bf16_t* __restrict__ dst, int Kdim, int KP){
  int idx = blockIdx.x*blockDim.x + threadIdx.x;
  int total = (KP/32)*4*32;
  if (idx >= total) return;
  int lane = idx & 31; int t = idx >> 5; int nt = t & 3; int chunk = t >> 2;
  int n = nt*16 + (lane & 15); int g = lane >> 4;
  bf16_t* o = dst + (size_t)idx*16;
  for (int j=0;j<16;++j){
    int k = chunk*32 + (j>>3)*16 + g*8 + (j&7);
    float v = (k < Kdim) ? W[(size_t)k*64 + n] : 0.f;
    o[j] = f2bf(v);
  }
}

// ---------------- setup kernels ----------------
__global__ void k_proj(const float* __restrict__ H, const float* __restrict__ Wp,
                       const float* __restrict__ bp, float* __restrict__ h,
                       float* __restrict__ cw, float* __restrict__ csum, int N){
  int n = blockIdx.x; int j = threadIdx.x;
  __shared__ float wv[NCHN];
  if (j < 68){
    float a = bp[j];
    const float* Hr = H + (size_t)n*64;
    for (int k=0;k<64;++k) a += Hr[k]*Wp[(size_t)k*68 + j];
    a = siluf_(a);
    if (j < 64) h[(size_t)n*64 + j] = a;
    else wv[j-64] = a;
  }
  __syncthreads();
  if (j == 0){
    int best = 0; float bv = wv[0];
    for (int c=1;c<NCHN;++c) if (wv[c] > bv){ bv = wv[c]; best = c; }
    for (int c=0;c<NCHN;++c) cw[n*NCHN+c] = (c==best)?1.f:0.f;
    csum[n] = 1.f;  // one-hot -> exactly one nonzero channel
  }
}

__global__ void k_attr(const float* __restrict__ X, const int* __restrict__ A,
                       const int* __restrict__ AP, const float* __restrict__ Et,
                       const float* __restrict__ Ep, float* __restrict__ attr, int N){
  int idx = blockIdx.x*blockDim.x + threadIdx.x;
  if (idx >= N*NCHN*ATTRD) return;
  int a = idx % ATTRD; int t = idx / ATTRD; int c = t % NCHN; int n = t / NCHN;
  float v;
  if (a < 3) v = X[(size_t)n*12 + c*3 + a];
  else { int e=a-3; v = Et[A[n*NCHN+c]*32 + e] + Ep[AP[n*NCHN+c]*32 + e]; }
  attr[idx] = v;
}

__global__ void k_bstart(const int* __restrict__ bid, const int* __restrict__ Bp,
                         int* __restrict__ start, int N){
  int n = blockIdx.x*blockDim.x + threadIdx.x;
  if (n == 0){ start[bid[0]] = 0; start[Bp[0]] = N; }
  if (n > 0 && n < N && bid[n] != bid[n-1]) start[bid[n]] = n;
}

__global__ void k_centers(const float* __restrict__ X, const int* __restrict__ bid,
                          const int* __restrict__ Bp, float* __restrict__ center, int N){
  __shared__ float s0[256], s1[256], s2[256], scn[256];
  int t = threadIdx.x; int B = Bp[0];
  for (int b=0;b<B;++b){
    float x=0,y=0,z=0,c=0;
    for (int n=t;n<N;n+=256) if (bid[n]==b){ x+=X[n*12+3]; y+=X[n*12+4]; z+=X[n*12+5]; c+=1.f; }
    s0[t]=x;s1[t]=y;s2[t]=z;scn[t]=c; __syncthreads();
    for (int o=128;o>0;o>>=1){
      if (t<o){ s0[t]+=s0[t+o]; s1[t]+=s1[t+o]; s2[t]+=s2[t+o]; scn[t]+=scn[t+o]; }
      __syncthreads();
    }
    if (t==0){ float cc = scn[0]<1.f?1.f:scn[0];
      center[b*3+0]=s0[0]/cc; center[b*3+1]=s1[0]/cc; center[b*3+2]=s2[0]/cc; }
    __syncthreads();
  }
}

__global__ void k_x0(const float* __restrict__ X, const int* __restrict__ bid,
                     const float* __restrict__ center, float* __restrict__ x, int N){
  int idx = blockIdx.x*blockDim.x+threadIdx.x;
  if (idx >= N*12) return;
  int n = idx/12; int r = idx%12; int k = r%3;
  x[idx] = X[idx] - center[bid[n]*3+k];
}

// ---------------- KNN (deterministic argmin rounds, reproduces top_k tie-breaks) ----------------
__global__ void k_knn(const float* __restrict__ X, const int* __restrict__ seg,
                      const int* __restrict__ bid, const int* __restrict__ start,
                      int* __restrict__ erow, int* __restrict__ ecol, int N){
  __shared__ float dist[MAXB];
  __shared__ float rv[256]; __shared__ int ri[256];
  int n = blockIdx.x; int t = threadIdx.x;
  int b = bid[n]; int s = start[b]; int e = start[b+1]; int m = e - s; if (m > MAXB) m = MAXB;
  int r = n - s;
  float xn0 = X[n*12+3], xn1 = X[n*12+4], xn2 = X[n*12+5];
  int sn = seg[n];
  for (int type=0; type<4; ++type){
    int k, ob;
    if (type==0){ k=30; ob = s*EPN + r*30; }
    else if (type==1){ k=10; ob = s*EPN + m*30 + r*10; }
    else if (type==2){ k=15; ob = s*EPN + m*40 + r*15; }
    else            { k=15; ob = s*EPN + m*55 + r*15; }
    for (int j=t;j<m;j+=256){
      int jg = s + j; int sj = seg[jg];
      bool mask;
      if (type==0) mask = (sj==sn);
      else if (type==1) mask = (sj!=sn);
      else if (type==2) mask = (sn==0)&&(sj>0);
      else             mask = (sn>0)&&(sj==0);
      float d = 1e9f;
      if (mask){
        float d0=xn0-X[jg*12+3], d1=xn1-X[jg*12+4], d2=xn2-X[jg*12+5];
        d = sqrtf(d0*d0+d1*d1+d2*d2);
      }
      dist[j]=d;
    }
    __syncthreads();
    for (int kk=0;kk<k;++kk){
      float bv = 3.4e38f; int bi = 0x7fffffff;
      for (int j=t;j<m;j+=256){
        float v = dist[j];
        if (v < bv || (v==bv && j<bi)){ bv=v; bi=j; }
      }
      rv[t]=bv; ri[t]=bi; __syncthreads();
      for (int o=128;o>0;o>>=1){
        if (t<o && (rv[t+o]<rv[t] || (rv[t+o]==rv[t] && ri[t+o]<ri[t]))){
          rv[t]=rv[t+o]; ri[t]=ri[t+o];
        }
        __syncthreads();
      }
      if (t==0){
        int w = ri[0];
        erow[ob+kk] = n; ecol[ob+kk] = s + w;
        dist[w] = 3.5e38f;
      }
      __syncthreads();
    }
    __syncthreads();
  }
}

__global__ void k_classify(const int* __restrict__ erow, const int* __restrict__ ecol,
                           const int* __restrict__ seg, int* __restrict__ tag, int E){
  int e = blockIdx.x*blockDim.x+threadIdx.x; if (e>=E) return;
  int rs = seg[erow[e]], cs = seg[ecol[e]];
  int t;
  if (rs>0 && cs>0) t=0; else if ((rs==0)!=(cs==0)) t=1; else t=2;
  tag[e]=t;
}

// ---------------- deterministic CSR build ----------------
__global__ void k_count(const int* __restrict__ key, const int* __restrict__ tag, int T,
                        int* __restrict__ cnt, int E){
  int e = blockIdx.x*blockDim.x+threadIdx.x; if (e>=E) return;
  if (tag[e]==T) atomicAdd(&cnt[key[e]], 1);
}
__global__ void k_scan(const int* __restrict__ cnt, int* __restrict__ off, int N){
  if (blockIdx.x==0 && threadIdx.x==0){
    int a=0; for (int i=0;i<N;++i){ off[i]=a; a+=cnt[i]; } off[N]=a;
  }
}
__global__ void k_fill(const int* __restrict__ key, const int* __restrict__ tag, int T,
                       const int* __restrict__ off, int* __restrict__ cur,
                       int* __restrict__ ids, int E){
  int e = blockIdx.x*blockDim.x+threadIdx.x; if (e>=E) return;
  if (tag[e]==T){ int k=key[e]; int p = off[k] + atomicAdd(&cur[k],1); ids[p]=e; }
}
__global__ void k_sortb(const int* __restrict__ off, int* __restrict__ ids, int N){
  int n = blockIdx.x*blockDim.x+threadIdx.x; if (n>=N) return;
  int s=off[n], e=off[n+1];
  for (int i=s+1;i<e;++i){ int v=ids[i]; int j=i-1;
    while(j>=s && ids[j]>v){ ids[j+1]=ids[j]; --j; } ids[j+1]=v; }
}

__global__ void k_pooled(const float* __restrict__ x, const float* __restrict__ cw,
                         const float* __restrict__ csum, float* __restrict__ pooledc, int N){
  int idx = blockIdx.x*blockDim.x+threadIdx.x; if (idx>=N*3) return;
  int n = idx/3, k = idx%3;
  float a=0;
  for (int c=0;c<NCHN;++c){ float m=(cw[n*NCHN+c]!=0.f)?1.f:0.f; a += x[n*12+c*3+k]*m; }
  float cs = csum[n]; if (cs<1.f) cs=1.f;
  pooledc[idx] = a/cs;
}

// ---------------- fused edge kernel: radial + edge MLP chain (WMMA) ----------------
struct GclArgs {
  const int* elist; const int* ecnt;
  const int* erow; const int* ecol;
  const float* x; const float* h; const float* attr; const float* cw; const float* csum;
  const float* pooledc;
  const bf16_t* wRad; const bf16_t* wE1; const bf16_t* wE2; const bf16_t* wC1;
  const float* bRad; const float* bE1; const float* bE2;
  const float* attW; const float* attB;
  const float* c1B; const float* wC2; const float* c2B;
  float* ef; float* contrib;
};

struct EdgeShared {
  alignas(16) bf16_t A[16*RADKP];   // radial A-tile; reused for e1/e2/c1 stages
  float T[16][NCHN*ATTRD];
  float msg[16][16];
  float fbuf[16][64];
  float sq[16][8];
  float rn[16];
  float attv[16];
  float ch[16][NCHN];
  int   row[16], col[16], eid[16];
  int   nact;
};

__launch_bounds__(128)
__global__ void k_edge(GclArgs A_){
  __shared__ EdgeShared S;
  extern __shared__ bf16_t dynW[];   // [WE1 | WE2 | WC1] B-fragments, async-staged
  int tid = threadIdx.x; int lane = tid & 31; int wid = tid >> 5;
  int E = *A_.ecnt;
  int base = blockIdx.x * 16;
  if (base >= E) return;

  // kick off async weight staging (hidden under the radial-tile build)
  bf16_t* sE1 = dynW;
  bf16_t* sE2 = dynW + WE1_ELEMS;
  bf16_t* sC1 = dynW + WE1_ELEMS + WE2_ELEMS;
  for (int t2 = tid; t2 < WSTAGE_ELEMS/8; t2 += 128){
    int el = t2*8;
    const bf16_t* src;
    if (el < WE1_ELEMS)                      src = A_.wE1 + el;
    else if (el < WE1_ELEMS + WE2_ELEMS)     src = A_.wE2 + (el - WE1_ELEMS);
    else                                     src = A_.wC1 + (el - WE1_ELEMS - WE2_ELEMS);
    async_copy_b128(dynW + el, src);
  }

  if (tid == 0){ int na = E - base; S.nact = na > 16 ? 16 : na; }
  if (tid < 16){
    int i = tid;
    if (base + i < E){
      int e = A_.elist[base+i];
      S.eid[i]=e; S.row[i]=A_.erow[e]; S.col[i]=A_.ecol[e];
    } else { S.eid[i]=-1; S.row[i]=0; S.col[i]=0; }
  }
  __syncthreads();
  int nact = S.nact;
  int i = tid >> 3, sub = tid & 7;     // 8 threads per edge slot
  // msg[c][d] = ||x_row[c]-x_col[d]|| * cw_row[c]*cw_col[d]
  {
    int row=S.row[i], col=S.col[i];
    for (int mm=sub; mm<16; mm+=8){
      int c=mm>>2, d=mm&3;
      float d0 = A_.x[row*12+c*3+0]-A_.x[col*12+d*3+0];
      float d1 = A_.x[row*12+c*3+1]-A_.x[col*12+d*3+1];
      float d2 = A_.x[row*12+c*3+2]-A_.x[col*12+d*3+2];
      float v = sqrtf(d0*d0+d1*d1+d2*d2 + 1e-12f);
      S.msg[i][mm] = v * A_.cw[row*NCHN+c]*A_.cw[col*NCHN+d];
    }
  }
  __syncthreads();
  // T[d][a] = sum_c P[c][a]*msg[c][d]
  {
    int row=S.row[i];
    const float* P = A_.attr + (size_t)row*140;
    for (int idx=sub; idx<140; idx+=8){
      int d = idx/ATTRD, a = idx%ATTRD;
      float s = 0;
      for (int c=0;c<NCHN;++c) s += P[c*ATTRD+a]*S.msg[i][c*4+d];
      S.T[i][idx]=s;
    }
  }
  __syncthreads();
  // radial[a][b] = sum_d T[d][a]*Q[d][b]  ->  bf16 A-tile + sumsq
  {
    float ssq = 0.f;
    if (i < nact){
      int col = S.col[i];
      const float* Q = A_.attr + (size_t)col*140;
      for (int b=sub;b<ATTRD;b+=8){
        float q0=Q[0*ATTRD+b], q1=Q[1*ATTRD+b], q2=Q[2*ATTRD+b], q3=Q[3*ATTRD+b];
        for (int a=0;a<ATTRD;++a){
          float v = S.T[i][0*ATTRD+a]*q0 + S.T[i][1*ATTRD+a]*q1
                  + S.T[i][2*ATTRD+a]*q2 + S.T[i][3*ATTRD+a]*q3;
          S.A[i*RADKP + a*ATTRD + b] = f2bf(v);
          ssq += v*v;
        }
      }
      for (int p=RADK+sub; p<RADKP; p+=8) S.A[i*RADKP+p] = f2bf(0.f);
    } else {
      for (int p=sub; p<RADKP; p+=8) S.A[i*RADKP+p] = f2bf(0.f);
    }
    S.sq[i][sub] = ssq;
  }
  __syncthreads();
  if (tid < 16){
    float s=0; for (int j2=0;j2<8;++j2) s += S.sq[tid][j2];
    S.rn[tid] = sqrtf(s + 1e-12f) + 1.f;
  }
  __syncthreads();
  // radial linear 1248->64 (39 WMMA chunks per wave, streamed from global + prefetch)
  v8f accR = gemm16(S.A, A_.wRad, RADKP, wid, lane);
  __syncthreads();
  // e1 input tile [16][192] = concat(h[row], h[col], (radial + bias)/rn)
  for (int t2=tid; t2<16*128; t2+=128){
    int ii=t2>>7, cc=t2&127;
    float v=0.f;
    if (ii < nact) v = (cc<64)? A_.h[(size_t)S.row[ii]*64+cc]
                              : A_.h[(size_t)S.col[ii]*64+(cc-64)];
    S.A[ii*192+cc] = f2bf(v);
  }
  {
    int g = lane>>4, nn = (lane&15) + wid*16;
    for (int r2=0;r2<8;++r2){
      int mm = r2 + 8*g;
      float v = (accR[r2] + A_.bRad[nn]) / S.rn[mm];
      S.A[mm*192 + 128 + nn] = f2bf(v);
    }
  }
  wait_async0();                 // staged weights resident before first consumer
  __syncthreads();
  v8f acc1 = gemm16(S.A, sE1, 192, wid, lane);
  __syncthreads();
  { int g=lane>>4, nn=(lane&15)+wid*16;
    for (int r2=0;r2<8;++r2){ int mm=r2+8*g;
      S.A[mm*64+nn] = f2bf(siluf_(acc1[r2] + A_.bE1[nn])); } }
  __syncthreads();
  v8f acc2 = gemm16(S.A, sE2, 64, wid, lane);
  __syncthreads();
  { int g=lane>>4, nn=(lane&15)+wid*16;
    for (int r2=0;r2<8;++r2){ int mm=r2+8*g;
      S.fbuf[mm][nn] = siluf_(acc2[r2] + A_.bE2[nn]); } }
  __syncthreads();
  if (tid < 16){
    float s = A_.attB[0];
    for (int k2=0;k2<64;++k2) s += S.fbuf[tid][k2]*A_.attW[k2];
    S.attv[tid] = sigmoidf_(s);
  }
  __syncthreads();
  // ef = e * att -> global f32 + bf16 restage for c1
  for (int t2=tid; t2<16*64; t2+=128){
    int ii=t2>>6, nn=t2&63;
    float v = S.fbuf[ii][nn]*S.attv[ii];
    S.A[ii*64+nn] = f2bf(v);
    if (ii < nact) A_.ef[(size_t)S.eid[ii]*64+nn] = v;
  }
  __syncthreads();
  v8f acc3 = gemm16(S.A, sC1, 64, wid, lane);
  __syncthreads();
  { int g=lane>>4, nn=(lane&15)+wid*16;
    for (int r2=0;r2<8;++r2){ int mm=r2+8*g;
      S.fbuf[mm][nn] = siluf_(acc3[r2] + A_.c1B[nn]); } }
  __syncthreads();
  if (tid < 64){
    int ii=tid>>2, d=tid&3;
    float s = A_.c2B[d];
    for (int k2=0;k2<64;++k2) s += S.fbuf[ii][k2]*A_.wC2[k2*4+d];
    S.ch[ii][d] = s;
  }
  __syncthreads();
  if (tid < 64){
    int ii=tid>>2, c=tid&3;
    if (ii < nact){
      int row=S.row[ii], col=S.col[ii];
      int ts = (int)(A_.csum[row] - 0.5f); if (ts<0) ts=0; if (ts>3) ts=3; // csum-1, rounded
      float inv = 1.f/(float)(4-ts);
      float pl = 0.f;
      for (int d=0; d<4; ++d) if (d>=c && d < c + (4-ts)) pl += S.ch[ii][d];
      pl *= inv;
      for (int k2=0;k2<3;++k2){
        float cd = A_.x[row*12+c*3+k2] - A_.pooledc[col*3+k2];
        A_.contrib[(size_t)S.eid[ii]*12 + c*3 + k2] = cd*pl;
      }
    }
  }
}

// ---------------- scatter means (deterministic CSR order) ----------------
__global__ void k_scatter_x(const int* __restrict__ off, const int* __restrict__ ids,
                            const float* __restrict__ contrib, float* __restrict__ x, int N){
  int n = blockIdx.x; int f = threadIdx.x;
  if (f >= 12) return;
  int s=off[n], e=off[n+1];
  float a=0;
  for (int i=s;i<e;++i) a += contrib[(size_t)ids[i]*12+f];
  int c = e-s; if (c<1) c=1;
  x[(size_t)n*12+f] += a/(float)c;
}
__global__ void k_scatter_agg(const int* __restrict__ off, const int* __restrict__ ids,
                              const float* __restrict__ ef, float* __restrict__ agg, int N){
  int n = blockIdx.x; int f = threadIdx.x;
  int s=off[n], e=off[n+1];
  float a=0;
  for (int i=s;i<e;++i) a += ef[(size_t)ids[i]*64+f];
  int c=e-s; if (c<1) c=1;
  agg[(size_t)n*64+f] = a/(float)c;
}

// ---------------- node MLP + layernorm (WMMA) ----------------
struct NodeArgs {
  const float* h; const float* agg;
  const bf16_t* wN1; const bf16_t* wN2;
  const float* bN1; const float* bN2; const float* g; const float* be;
  float* out; int N;
};
__launch_bounds__(128)
__global__ void k_node(NodeArgs P){
  __shared__ alignas(16) bf16_t A[16*128];
  __shared__ float y[16][64];
  __shared__ float mu[16], rs[16];
  int tid=threadIdx.x, lane=tid&31, wid=tid>>5;
  int n0 = blockIdx.x*16;
  for (int t=tid;t<16*128;t+=128){
    int m=t>>7, c=t&127; int n=n0+m;
    float v = 0.f;
    if (n < P.N) v = (c<64)? P.h[(size_t)n*64+c] : P.agg[(size_t)n*64+(c-64)];
    A[m*128+c]=f2bf(v);
  }
  __syncthreads();
  v8f a1 = gemm16(A, P.wN1, 128, wid, lane);
  __syncthreads();
  { int g=lane>>4, nn=(lane&15)+wid*16;
    for (int r=0;r<8;++r){ int m=r+8*g; A[m*64+nn]=f2bf(siluf_(a1[r]+P.bN1[nn])); } }
  __syncthreads();
  v8f a2 = gemm16(A, P.wN2, 64, wid, lane);
  { int g=lane>>4, nn=(lane&15)+wid*16;
    for (int r=0;r<8;++r){ int m=r+8*g; int n=n0+m;
      float hv = (n<P.N)? P.h[(size_t)n*64+nn] : 0.f;
      y[m][nn] = a2[r]+P.bN2[nn]+hv; } }
  __syncthreads();
  if (tid<16){
    float s=0; for (int k=0;k<64;++k) s+=y[tid][k]; float m_=s/64.f;
    float v=0; for (int k=0;k<64;++k){ float d=y[tid][k]-m_; v+=d*d; }
    mu[tid]=m_; rs[tid]=rsqrtf(v/64.f + 1e-5f);
  }
  __syncthreads();
  for (int t=tid;t<16*64;t+=128){
    int m=t>>6, nn=t&63; int n=n0+m;
    if (n < P.N)
      P.out[(size_t)n*64+nn] = (y[m][nn]-mu[m])*rs[m]*P.g[nn]+P.be[nn];
  }
}

// ---------------- gate (WMMA) ----------------
__launch_bounds__(128)
__global__ void k_gate(const float* __restrict__ hc, const float* __restrict__ hi,
                       const bf16_t* __restrict__ wG, const float* __restrict__ bG,
                       float* __restrict__ h, int N){
  __shared__ alignas(16) bf16_t A[16*128];
  int tid=threadIdx.x, lane=tid&31, wid=tid>>5;
  int n0=blockIdx.x*16;
  for (int t=tid;t<16*128;t+=128){
    int m=t>>7, c=t&127; int n=n0+m;
    float v=0.f;
    if (n < N) v=(c<64)? hc[(size_t)n*64+c] : hi[(size_t)n*64+(c-64)];
    A[m*128+c]=f2bf(v);
  }
  __syncthreads();
  v8f a = gemm16(A, wG, 128, wid, lane);
  int g=lane>>4, nn=(lane&15)+wid*16;
  for (int r=0;r<8;++r){
    int m=r+8*g; int n=n0+m;
    if (n < N){
      float gd=sigmoidf_(a[r]+bG[nn]);
      h[(size_t)n*64+nn]=gd*hc[(size_t)n*64+nn]+(1.f-gd)*hi[(size_t)n*64+nn];
    }
  }
}

// ---------------- host orchestration ----------------
extern "C" void kernel_launch(void* const* d_in, const int* in_sizes, int n_in,
                              void* d_out, int out_size, void* d_ws, size_t ws_size,
                              hipStream_t stream){
  (void)n_in; (void)out_size; (void)ws_size;
  auto F=[&](int i){ return (const float*)d_in[i]; };
  auto I=[&](int i){ return (const int*)d_in[i]; };
  const float* X=F(0); const float* H=F(1);
  const int* A=I(2); const int* AP=I(3); const int* seg=I(4); const int* bid=I(5);
  const int* Bp=I(6);
  int N = in_sizes[4];
  int E = N*EPN;
  // params: jax pytree order (sorted dict keys; lists in order; lin = {b,w})
  const float* Ep=F(8); const float* Et=F(9);
  int LB[4] = {10, 26, 44, 60};               // ctx0, ctx1, inter0, inter1
  const float* gateB=F(42); const float* gateW=F(43);
  const float* projB=F(76); const float* projW=F(77);
  const float* radB=F(78);  const float* radW=F(79);

  char* wsb=(char*)d_ws; size_t off=0;
  auto alloc=[&](size_t bytes)->void*{ size_t a=(off+255)&~(size_t)255; off=a+bytes; return (void*)(wsb+a); };
  float* hB   =(float*)alloc((size_t)N*64*4);
  float* hcB  =(float*)alloc((size_t)N*64*4);
  float* hiB  =(float*)alloc((size_t)N*64*4);
  float* aggB =(float*)alloc((size_t)N*64*4);
  float* xB   =(float*)alloc((size_t)N*12*4);
  float* attrB=(float*)alloc((size_t)N*140*4);
  float* cwB  =(float*)alloc((size_t)N*4*4);
  float* csumB=(float*)alloc((size_t)N*4);
  float* pooledB=(float*)alloc((size_t)N*3*4);
  float* centerB=(float*)alloc(64*3*4);
  int* bstart=(int*)alloc(66*4);
  int* erow=(int*)alloc((size_t)E*4);
  int* ecol=(int*)alloc((size_t)E*4);
  int* etag=(int*)alloc((size_t)E*4);
  int *rowOff[2], *rowCur[2], *rowIds[2], *colOff[2], *colCur[2], *colIds[2];
  for (int s2=0;s2<2;++s2){
    rowOff[s2]=(int*)alloc((size_t)(N+1)*4); rowCur[s2]=(int*)alloc((size_t)N*4);
    rowIds[s2]=(int*)alloc((size_t)E*4);
    colOff[s2]=(int*)alloc((size_t)(N+1)*4); colCur[s2]=(int*)alloc((size_t)N*4);
    colIds[s2]=(int*)alloc((size_t)E*4);
  }
  float* efB     =(float*)alloc((size_t)E*64*4);
  float* contribB=(float*)alloc((size_t)E*12*4);
  int* cntTmp=(int*)alloc((size_t)N*4);
  bf16_t* pRad =(bf16_t*)alloc((size_t)RADKP*64*2);
  bf16_t* pGate=(bf16_t*)alloc((size_t)128*64*2);
  bf16_t *pE1[4], *pE2[4], *pC1[4], *pN1[4], *pN2[4];
  for (int l=0;l<4;++l){
    pE1[l]=(bf16_t*)alloc((size_t)192*64*2);
    pE2[l]=(bf16_t*)alloc((size_t)64*64*2);
    pC1[l]=(bf16_t*)alloc((size_t)64*64*2);
    pN1[l]=(bf16_t*)alloc((size_t)128*64*2);
    pN2[l]=(bf16_t*)alloc((size_t)64*64*2);
  }

  auto pack=[&](const float* W, bf16_t* dst, int Kdim, int KP){
    int threads=(KP/32)*4*32;
    k_pack<<<dim3((threads+255)/256), dim3(256), 0, stream>>>(W, dst, Kdim, KP);
  };
  pack(radW,  pRad,  RADK, RADKP);
  pack(gateW, pGate, 128,  128);
  for (int l=0;l<4;++l){
    int b=LB[l];
    pack(F(b+8),  pE1[l], 192, 192);   // e1.w
    pack(F(b+10), pE2[l], 64,  64);    // e2.w
    pack(F(b+4),  pC1[l], 64,  64);    // c1.w
    pack(F(b+13), pN1[l], 128, 128);   // n1.w
    pack(F(b+15), pN2[l], 64,  64);    // n2.w
  }

  k_proj<<<N, 96, 0, stream>>>(H, projW, projB, hB, cwB, csumB, N);
  { int tot=N*NCHN*ATTRD; k_attr<<<(tot+255)/256,256,0,stream>>>(X,A,AP,Et,Ep,attrB,N); }
  k_bstart<<<(N+255)/256,256,0,stream>>>(bid,Bp,bstart,N);
  k_centers<<<1,256,0,stream>>>(X,bid,Bp,centerB,N);
  { int tot=N*12; k_x0<<<(tot+255)/256,256,0,stream>>>(X,bid,centerB,xB,N); }
  k_knn<<<N,256,0,stream>>>(X,seg,bid,bstart,erow,ecol,N);
  k_classify<<<(E+255)/256,256,0,stream>>>(erow,ecol,seg,etag,E);
  for (int s2=0;s2<2;++s2){
    hipMemsetAsync(cntTmp,0,(size_t)N*4,stream);
    k_count<<<(E+255)/256,256,0,stream>>>(erow,etag,s2,cntTmp,E);
    k_scan<<<1,1,0,stream>>>(cntTmp,rowOff[s2],N);
    hipMemsetAsync(rowCur[s2],0,(size_t)N*4,stream);
    k_fill<<<(E+255)/256,256,0,stream>>>(erow,etag,s2,rowOff[s2],rowCur[s2],rowIds[s2],E);
    k_sortb<<<(N+63)/64,64,0,stream>>>(rowOff[s2],rowIds[s2],N);
    hipMemsetAsync(cntTmp,0,(size_t)N*4,stream);
    k_count<<<(E+255)/256,256,0,stream>>>(ecol,etag,s2,cntTmp,E);
    k_scan<<<1,1,0,stream>>>(cntTmp,colOff[s2],N);
    hipMemsetAsync(colCur[s2],0,(size_t)N*4,stream);
    k_fill<<<(E+255)/256,256,0,stream>>>(ecol,etag,s2,colOff[s2],colCur[s2],colIds[s2],E);
    k_sortb<<<(N+63)/64,64,0,stream>>>(colOff[s2],colIds[s2],N);
  }

  int egrid = (E + 15)/16;
  int ngrid = (N + 15)/16;
  auto run_gcl=[&](int lidx, int set, float* outH){
    int b=LB[lidx];
    k_pooled<<<(N*3+255)/256,256,0,stream>>>(xB,cwB,csumB,pooledB,N);
    GclArgs ga;
    ga.elist=rowIds[set]; ga.ecnt=rowOff[set]+N;
    ga.erow=erow; ga.ecol=ecol;
    ga.x=xB; ga.h=hB; ga.attr=attrB; ga.cw=cwB; ga.csum=csumB; ga.pooledc=pooledB;
    ga.wRad=pRad; ga.wE1=pE1[lidx]; ga.wE2=pE2[lidx]; ga.wC1=pC1[lidx];
    ga.bRad=radB; ga.bE1=F(b+7); ga.bE2=F(b+9);
    ga.attW=F(b+1); ga.attB=F(b+0);
    ga.c1B=F(b+3); ga.wC2=F(b+6); ga.c2B=F(b+5);
    ga.ef=efB; ga.contrib=contribB;
    k_edge<<<egrid,128,WSTAGE_ELEMS*2,stream>>>(ga);
    k_scatter_x<<<N,32,0,stream>>>(rowOff[set],rowIds[set],contribB,xB,N);
    k_scatter_agg<<<N,64,0,stream>>>(colOff[set],colIds[set],efB,aggB,N);
    NodeArgs na{hB,aggB,pN1[lidx],pN2[lidx],F(b+12),F(b+14),F(b+11),F(b+2),outH,N};
    k_node<<<ngrid,128,0,stream>>>(na);
  };

  for (int li=0; li<2; ++li){
    run_gcl(li,   0, hcB);   // ctx layer on intra edges (updates x first)
    run_gcl(2+li, 1, hiB);   // inter layer on inter edges (sees updated x)
    k_gate<<<ngrid,128,0,stream>>>(hcB,hiB,pGate,gateB,hB,N);
  }

  hipMemcpyAsync(d_out, hB, (size_t)N*64*4, hipMemcpyDeviceToDevice, stream);
  hipMemcpyAsync((float*)d_out + (size_t)N*64, xB, (size_t)N*12*4,
                 hipMemcpyDeviceToDevice, stream);
}
